// Cluster3DVolume_35837207118528
// MI455X (gfx1250) — matrix operand order repaired and checked by reading, compiled, not run on hardware
//
#include <hip/hip_runtime.h>
#include <stdint.h>

typedef __attribute__((ext_vector_type(16))) _Float16 v16h;
typedef __attribute__((ext_vector_type(16))) __bf16   v16bf;
typedef __attribute__((ext_vector_type(8)))  float    v8f;

#define DEVFN static __device__ __forceinline__

constexpr int BATCH  = 2;
constexpr int CCH    = 256;
constexpr int DHW    = 131072;   // 32*64*64
constexpr int TCHUNK = 256;      // tokens per attention workgroup

#if defined(__has_builtin)
#  if __has_builtin(__builtin_amdgcn_tensor_load_to_lds)
#    define HAVE_TDM 1
#  endif
#endif
#ifndef HAVE_TDM
#  define HAVE_TDM 0
#endif

DEVFN v8f vzero8(){
  v8f z;
#pragma unroll
  for (int i = 0; i < 8; i++) z[i] = 0.f;
  return z;
}

// native RNE convert (v_cvt_*bf16_f32 on gfx1250) instead of manual bit ops
DEVFN __bf16 f2bf(float f){ return (__bf16)f; }

// token index n within octant-block b1 -> flat spatial index p in [0, DHW)
DEVFN int token_p(int b1, int n){
  int fw = b1 & 1, fh = (b1 >> 1) & 1, fd = (b1 >> 2) & 1;
  int wi = n & 31, hi_ = (n >> 5) & 31, di = n >> 10;
  return (fd*16 + di)*4096 + (fh*32 + hi_)*64 + (fw*32 + wi);
}

#if HAVE_TDM
typedef __attribute__((ext_vector_type(4))) unsigned v4u;
typedef __attribute__((ext_vector_type(8))) int      v8i;
typedef __attribute__((ext_vector_type(4))) int      v4i;

// 1-D TDM copy: `dwords` DWORDs from global -> LDS (one descriptor, count=1)
// 6-arg toolchain variant: (g0 u32x4, g1 i32x8, g2 i32x4, g3 i32x4, i32x8, cpol)
DEVFN void tdm_load_dwords(void* ldsDst, const void* gsrc, unsigned dwords){
  unsigned lds = (unsigned)(uintptr_t)ldsDst;           // flat addr low 32b == LDS offset
  unsigned long long ga = (unsigned long long)(uintptr_t)gsrc;
  v4u g0;
  g0[0] = 1u;                                           // count=1, user mode, no gather
  g0[1] = lds;                                          // lds_addr
  g0[2] = (unsigned)(ga & 0xffffffffu);                 // global_addr[31:0]
  g0[3] = (unsigned)((ga >> 32) & 0x01ffffffu) | (2u << 30);  // addr[56:32] | type=2
  v8i g1;
  g1[0] = (int)(2u << 16);                              // data_size = 4 bytes
  g1[1] = (int)((dwords & 0xffffu) << 16);              // tensor_dim0[15:0]
  g1[2] = (int)(((dwords >> 16) & 0xffffu) | (1u << 16)); // tensor_dim0[31:16] | tensor_dim1=1
  g1[3] = (int)((dwords & 0xffffu) << 16);              // tile_dim0
  g1[4] = 0;                                            // tile_dim1=0 (1-D), tile_dim2=0
  g1[5] = (int)dwords;                                  // tensor_dim0_stride[31:0]
  g1[6] = 0;
  g1[7] = 0;
  v4i z4; z4[0]=0; z4[1]=0; z4[2]=0; z4[3]=0;
  v8i z8;
#pragma unroll
  for (int i = 0; i < 8; i++) z8[i] = 0;
  __builtin_amdgcn_tensor_load_to_lds(g0, g1, z4, z4, z8, 0);
}
#endif

// stage nhalf f16 elements (nhalf/2 dwords) from global into LDS.
// Completion is published to the workgroup by the caller's next __syncthreads().
DEVFN void stage_tile(_Float16* ldsDst, const _Float16* gsrc, int nhalf,
                      int tid, int wave){
#if HAVE_TDM
  if (wave == 0){
    tdm_load_dwords(ldsDst, gsrc, (unsigned)(nhalf >> 1));
    __builtin_amdgcn_s_wait_tensorcnt(0);
  }
#else
  for (int i = tid; i < (nhalf >> 1); i += 256)
    ((unsigned*)ldsDst)[i] = ((const unsigned*)gsrc)[i];
#endif
}

// ---------------------------------------------------------------------------
// Kernel A: feat = f_w@x + f_b ; value = v_w@x + v_b  (bf16 WMMA, f32 accum)
// x: [b][c][p] channel-major;  outputs token-major [b][p][256]
// ---------------------------------------------------------------------------
__global__ void __launch_bounds__(256) k_qkv(
    const float* __restrict__ x,  const float* __restrict__ fw, const float* __restrict__ fb,
    const float* __restrict__ vw, const float* __restrict__ vb,
    float* __restrict__ featT, float* __restrict__ valT)
{
  extern __shared__ float smemA[];                 // 8 waves * 2048 floats
  const int tid  = threadIdx.x;
  const int wave = tid >> 5, lane = tid & 31;
  const int lo = lane & 15, hb = lane >> 4;
  const int b    = blockIdx.x >> 11;
  const int pblk = blockIdx.x & 2047;
  const int oh   = wave & 1;
  const int p0   = pblk * 64 + (wave >> 1) * 16;
  const float* xb = x + (size_t)b * CCH * DHW;

  v8f accF[8], accV[8];
#pragma unroll
  for (int t = 0; t < 8; t++){ accF[t] = vzero8(); accV[t] = vzero8(); }

  for (int ks = 0; ks < 8; ks++){
    // B tile: x[c][p]; element j -> c = ks*32 + hb*16 + j, p = p0 + lo
    v16bf bt;
    const int cb = ks*32 + hb*16;
    const float* xp = xb + p0 + lo;
#pragma unroll
    for (int j = 0; j < 16; j++)
      bt[j] = f2bf(xp[(size_t)(cb + j) * DHW]);
    if (ks < 7)
      __builtin_prefetch(xp + (size_t)(cb + 32) * DHW, 0, 1);   // global_prefetch_b8
    const int c0 = ks*32 + hb*16;
#pragma unroll
    for (int mt = 0; mt < 8; mt++){
      const int orow = oh*128 + mt*16 + lo;
      const float* fr = fw + (size_t)orow * CCH;
      const float* vr = vw + (size_t)orow * CCH;
      v16bf af, av;
#pragma unroll
      for (int j = 0; j < 8; j++){
        af[j]   = f2bf(fr[c0 + j]);      av[j]   = f2bf(vr[c0 + j]);
        af[j+8] = f2bf(fr[c0 + 8 + j]); av[j+8] = f2bf(vr[c0 + 8 + j]);
      }
      accF[mt] = __builtin_amdgcn_wmma_f32_16x16x32_bf16(false, af, false, bt, (short)0, accF[mt], false, false);
      accV[mt] = __builtin_amdgcn_wmma_f32_16x16x32_bf16(false, av, false, bt, (short)0, accV[mt], false, false);
    }
  }
  // bias + per-wave LDS transpose, then coalesced token-major float4 stores
  float* tb = smemA + wave * 2048;                 // [16 p][128 o]
#pragma unroll
  for (int pass = 0; pass < 2; pass++){
    const float* bias = pass ? vb : fb;
    float* dst = pass ? valT : featT;
#pragma unroll
    for (int mt = 0; mt < 8; mt++){
      v8f a = pass ? accV[mt] : accF[mt];
#pragma unroll
      for (int i = 0; i < 8; i++){
        int ol = mt*16 + hb*8 + i;
        tb[lo*128 + ol] = a[i] + bias[oh*128 + ol];
      }
    }
    for (int r = 0; r < 16; r++){
      float4 v4 = *(const float4*)(tb + r*128 + lane*4);
      *(float4*)(dst + ((size_t)b*DHW + p0 + r)*CCH + oh*128 + lane*4) = v4;
    }
  }
}

// ---------------------------------------------------------------------------
// Kernel B: adaptive-avg-pool centers, L2-normalized (f16) + value_centers
// ---------------------------------------------------------------------------
__global__ void __launch_bounds__(64) k_pool(
    const float* __restrict__ featT, const float* __restrict__ valT,
    _Float16* __restrict__ cnH, float* __restrict__ vcent)
{
  const int blk = blockIdx.x;
  const int b1 = blk >> 6, k = blk & 63;
  const int c = threadIdx.x;
  const int kd = k >> 4, kh = (k >> 2) & 3, kw = k & 3;
  const int e = (b1 >> 3) & 3, b = b1 >> 5;
  float fs = 0.f, vs = 0.f;
  for (int dd = 0; dd < 4; dd++)
    for (int hh = 0; hh < 8; hh++)
      for (int ww = 0; ww < 8; ww++){
        int n = (kd*4 + dd)*1024 + (kh*8 + hh)*32 + (kw*8 + ww);
        size_t base = ((size_t)b*DHW + token_p(b1, n))*CCH + e*64 + c;
        fs += featT[base];
        vs += valT[base];
      }
  fs *= (1.f/256.f); vs *= (1.f/256.f);
  __shared__ float cbuf[64];
  cbuf[c] = fs;
  __syncthreads();
  float ss = 0.f;
  for (int i = 0; i < 64; i++){ float t = cbuf[i]; ss += t*t; }
  float inv = 1.f / fmaxf(sqrtf(ss), 1e-12f);
  cnH[((size_t)b1*64 + k)*64 + c] = (_Float16)(fs * inv);
  vcent[((size_t)b1*64 + k)*64 + c] = vs;
}

// ---------------------------------------------------------------------------
// Shared attention front-end: normalize token tile, f16-WMMA cosine sim,
// leaky-relu, per-token top-8 masked softmax -> attn in LDS
// `cn` points at the TDM-staged centers tile in LDS.
// ---------------------------------------------------------------------------
template<bool TRANS>
DEVFN void attn_front(const float* __restrict__ featT, const _Float16* cn,
                      float alpha, float beta, int b1, int n0,
                      _Float16* tn, float* sim, _Float16* attn)
{
  const int tid = threadIdx.x, wave = tid >> 5, lane = tid & 31;
  const int lo = lane & 15, hb = lane >> 4;
  const int e = (b1 >> 3) & 3, b = b1 >> 5;
  {
    const int t = tid;
    const float4* src = (const float4*)(featT + ((size_t)b*DHW + token_p(b1, n0 + t))*CCH + e*64);
    float ss = 0.f;
#pragma unroll
    for (int i = 0; i < 16; i++){
      float4 v = src[i];
      ss += v.x*v.x + v.y*v.y + v.z*v.z + v.w*v.w;
    }
    float inv = 1.f / fmaxf(sqrtf(ss), 1e-12f);
#pragma unroll
    for (int i = 0; i < 16; i++){
      float4 v = src[i];
      tn[t*64 + 4*i+0] = (_Float16)(v.x*inv);
      tn[t*64 + 4*i+1] = (_Float16)(v.y*inv);
      tn[t*64 + 4*i+2] = (_Float16)(v.z*inv);
      tn[t*64 + 4*i+3] = (_Float16)(v.w*inv);
    }
  }
  __syncthreads();      // tokens + TDM-staged tiles now visible to all waves
  // sim[k][n] = leaky_relu(beta + alpha * cn_k . tn_n)
  for (int nn = 0; nn < 2; nn++){
    const int nt = wave*2 + nn;
#pragma unroll
    for (int kt = 0; kt < 4; kt++){
      v8f acc = vzero8();
#pragma unroll
      for (int cs = 0; cs < 2; cs++){
        v16h a, bm;
        const _Float16* cr = cn + (kt*16 + lo)*64;
        const int c0 = cs*32 + hb*16;
#pragma unroll
        for (int j = 0; j < 8; j++){ a[j] = cr[c0 + j]; a[j+8] = cr[c0 + 8 + j]; }
        const int cb = cs*32 + hb*16;
        const _Float16* tr = tn + (nt*16 + lo)*64;
#pragma unroll
        for (int j = 0; j < 16; j++) bm[j] = tr[cb + j];
        acc = __builtin_amdgcn_wmma_f32_16x16x32_f16(false, a, false, bm, (short)0, acc, false, false);
      }
#pragma unroll
      for (int i = 0; i < 8; i++){
        float s = beta + alpha * acc[i];
        s = s > 0.f ? s : 0.2f*s;
        sim[(kt*16 + hb*8 + i)*TCHUNK + nt*16 + lo] = s;
      }
    }
  }
  __syncthreads();
  // per-token top-8 + softmax over selected clusters (others exactly 0)
  {
    const int t = tid;
#pragma unroll
    for (int k = 0; k < 64; k++){
      if (TRANS) attn[t*64 + k] = (_Float16)0.f;
      else       attn[k*TCHUNK + t] = (_Float16)0.f;
    }
    float m = 0.f, Z = 0.f, ev[8]; int idx[8];
#pragma unroll
    for (int j = 0; j < 8; j++){
      float best = -3.0e38f; int bk = 0;
      for (int k = 0; k < 64; k++){
        float v = sim[k*TCHUNK + t];
        if (v > best){ best = v; bk = k; }
      }
      sim[bk*TCHUNK + t] = -3.0e38f;
      if (j == 0) m = best;
      float e2 = __expf(best - m);
      ev[j] = e2; Z += e2; idx[j] = bk;
    }
    const float invZ = 1.f / Z;
#pragma unroll
    for (int j = 0; j < 8; j++){
      _Float16 a2 = (_Float16)(ev[j] * invZ);
      if (TRANS) attn[t*64 + idx[j]] = a2;
      else       attn[idx[j]*TCHUNK + t] = a2;
    }
  }
  __syncthreads();
}

// ---------------------------------------------------------------------------
// Kernel C (pass A): cluster_sum[k][c] += attn @ value ; denom[k] += row-sum
// ---------------------------------------------------------------------------
__global__ void __launch_bounds__(256) k_passA(
    const float* __restrict__ featT, const float* __restrict__ valT,
    const _Float16* __restrict__ cnH,
    const float* __restrict__ alphap, const float* __restrict__ betap,
    float* __restrict__ csum, float* __restrict__ denom)
{
  extern __shared__ char smem[];
  _Float16* tn   = (_Float16*)smem;                 // 256*64 f16
  _Float16* va   = tn + TCHUNK*64;                  // 256*64 f16
  float*    sim  = (float*)(va + TCHUNK*64);        // 64*256 f32
  _Float16* attn = (_Float16*)(sim + 64*TCHUNK);    // 64*256 f16
  _Float16* cnL  = attn + 64*TCHUNK;                // 64*64  f16 (TDM-staged)
  const int b1 = blockIdx.x >> 6;
  const int n0 = (blockIdx.x & 63) * TCHUNK;
  const int tid = threadIdx.x, wave = tid >> 5, lane = tid & 31;
  const int lo = lane & 15, hb = lane >> 4;
  const int e = (b1 >> 3) & 3, b = b1 >> 5;

  stage_tile(cnL, cnH + (size_t)b1*4096, 4096, tid, wave);   // 8 KB via TDM
  {
    const int t = tid;
    const float4* src = (const float4*)(valT + ((size_t)b*DHW + token_p(b1, n0 + t))*CCH + e*64);
#pragma unroll
    for (int i = 0; i < 16; i++){
      float4 v = src[i];
      va[t*64 + 4*i+0] = (_Float16)v.x;
      va[t*64 + 4*i+1] = (_Float16)v.y;
      va[t*64 + 4*i+2] = (_Float16)v.z;
      va[t*64 + 4*i+3] = (_Float16)v.w;
    }
  }
  attn_front<false>(featT, cnL, alphap[0], betap[0], b1, n0, tn, sim, attn);
  if (tid < 64){
    float s = 0.f;
    for (int n = 0; n < TCHUNK; n++) s += (float)attn[tid*TCHUNK + n];
    atomicAdd(&denom[b1*64 + tid], s);
  }
  // partial cluster_sum = attn[64x256] @ value[256x64] via f16 WMMA
#pragma unroll
  for (int tt = 0; tt < 2; tt++){
    const int tile = wave*2 + tt, kt = tile >> 2, ct = tile & 3;
    v8f acc = vzero8();
#pragma unroll
    for (int ns = 0; ns < 8; ns++){
      v16h a, bm;
      const _Float16* ar = attn + (kt*16 + lo)*TCHUNK;
      const int nb0 = ns*32 + hb*16;
#pragma unroll
      for (int j = 0; j < 8; j++){ a[j] = ar[nb0 + j]; a[j+8] = ar[nb0 + 8 + j]; }
      const int nb = ns*32 + hb*16;
#pragma unroll
      for (int j = 0; j < 16; j++) bm[j] = va[(nb + j)*64 + ct*16 + lo];
      acc = __builtin_amdgcn_wmma_f32_16x16x32_f16(false, a, false, bm, (short)0, acc, false, false);
    }
#pragma unroll
    for (int i = 0; i < 8; i++)
      atomicAdd(&csum[((size_t)b1*64 + kt*16 + hb*8 + i)*64 + ct*16 + lo], acc[i]);
  }
}

// ---------------------------------------------------------------------------
// Kernel D: cluster_feat = csum / max(denom,1e-6) + value_centers  (f16)
// ---------------------------------------------------------------------------
__global__ void __launch_bounds__(256) k_cfeat(
    const float* __restrict__ csum, const float* __restrict__ denom,
    const float* __restrict__ vcent, _Float16* __restrict__ cfeatH)
{
  int i = blockIdx.x * 256 + threadIdx.x;          // 262144 total
  int kk = i >> 6;
  float d = fmaxf(denom[kk], 1e-6f);
  cfeatH[i] = (_Float16)(csum[i] / d + vcent[i]);
}

// ---------------------------------------------------------------------------
// Kernel E (pass B): out_tokens = attn^T @ cluster_feat  (recompute attn)
// ---------------------------------------------------------------------------
__global__ void __launch_bounds__(256) k_passB(
    const float* __restrict__ featT, const _Float16* __restrict__ cnH,
    const _Float16* __restrict__ cfeatH,
    const float* __restrict__ alphap, const float* __restrict__ betap,
    float* __restrict__ outT)
{
  extern __shared__ char smem[];
  _Float16* tn    = (_Float16*)smem;               // 256*64
  float*    sim   = (float*)(tn + TCHUNK*64);      // 64*256
  _Float16* attnT = (_Float16*)(sim + 64*TCHUNK);  // 256*64 (token-major)
  _Float16* cnL   = attnT + TCHUNK*64;             // 64*64 (TDM-staged)
  _Float16* cfL   = cnL + 64*64;                   // 64*64 (TDM-staged)
  const int b1 = blockIdx.x >> 6;
  const int n0 = (blockIdx.x & 63) * TCHUNK;
  const int tid = threadIdx.x, wave = tid >> 5, lane = tid & 31;
  const int lo = lane & 15, hb = lane >> 4;
  const int e = (b1 >> 3) & 3, b = b1 >> 5;

  stage_tile(cnL, cnH    + (size_t)b1*4096, 4096, tid, wave);
  stage_tile(cfL, cfeatH + (size_t)b1*4096, 4096, tid, wave);

  attn_front<true>(featT, cnL, alphap[0], betap[0], b1, n0, tn, sim, attnT);
  for (int tt = 0; tt < 8; tt++){
    const int tile = wave*8 + tt, mt = tile >> 2, ct = tile & 3;
    v8f acc = vzero8();
#pragma unroll
    for (int ks = 0; ks < 2; ks++){
      v16h a, bm;
      const _Float16* ar = attnT + (mt*16 + lo)*64;
      const int k0 = ks*32 + hb*16;
#pragma unroll
      for (int j = 0; j < 8; j++){ a[j] = ar[k0 + j]; a[j+8] = ar[k0 + 8 + j]; }
      const int kb = ks*32 + hb*16;
#pragma unroll
      for (int j = 0; j < 16; j++) bm[j] = cfL[(kb + j)*64 + ct*16 + lo];
      acc = __builtin_amdgcn_wmma_f32_16x16x32_f16(false, a, false, bm, (short)0, acc, false, false);
    }
#pragma unroll
    for (int i = 0; i < 8; i++){
      const int n = n0 + mt*16 + hb*8 + i;
      outT[((size_t)b*DHW + token_p(b1, n))*CCH + e*64 + ct*16 + lo] = acc[i];
    }
  }
}

// ---------------------------------------------------------------------------
// Kernel F: final projection out = p_w @ outT + p_b  -> [b][o][p] (d_out)
// ---------------------------------------------------------------------------
__global__ void __launch_bounds__(256) k_proj(
    const float* __restrict__ outT, const float* __restrict__ pw,
    const float* __restrict__ pb, float* __restrict__ out)
{
  const int tid = threadIdx.x, wave = tid >> 5, lane = tid & 31;
  const int lo = lane & 15, hb = lane >> 4;
  const int b = blockIdx.x >> 11, pblk = blockIdx.x & 2047;
  const int oh = wave & 1;
  const int p0 = pblk*64 + (wave >> 1)*16;
  v8f acc[8];
#pragma unroll
  for (int t = 0; t < 8; t++) acc[t] = vzero8();
  for (int ks = 0; ks < 8; ks++){
    v16bf bt;
    const float* op = outT + ((size_t)b*DHW + p0 + lo)*CCH + ks*32 + hb*16;
#pragma unroll
    for (int j = 0; j < 16; j++) bt[j] = f2bf(op[j]);
    const int c0 = ks*32 + hb*16;
#pragma unroll
    for (int mt = 0; mt < 8; mt++){
      const int orow = oh*128 + mt*16 + lo;
      const float* wr = pw + (size_t)orow*CCH;
      v16bf a;
#pragma unroll
      for (int j = 0; j < 8; j++){ a[j] = f2bf(wr[c0 + j]); a[j+8] = f2bf(wr[c0 + 8 + j]); }
      acc[mt] = __builtin_amdgcn_wmma_f32_16x16x32_bf16(false, a, false, bt, (short)0, acc[mt], false, false);
    }
  }
#pragma unroll
  for (int mt = 0; mt < 8; mt++){
#pragma unroll
    for (int i = 0; i < 8; i++){
      const int o = oh*128 + mt*16 + hb*8 + i;
      out[((size_t)b*CCH + o)*DHW + p0 + lo] = acc[mt][i] + pb[o];
    }
  }
}

// ---------------------------------------------------------------------------
extern "C" void kernel_launch(void* const* d_in, const int* in_sizes, int n_in,
                              void* d_out, int out_size, void* d_ws, size_t ws_size,
                              hipStream_t stream)
{
  const float* x     = (const float*)d_in[0];
  const float* fw    = (const float*)d_in[1];
  const float* fb    = (const float*)d_in[2];
  const float* vw    = (const float*)d_in[3];
  const float* vb    = (const float*)d_in[4];
  const float* pw    = (const float*)d_in[5];
  const float* pb    = (const float*)d_in[6];
  const float* alpha = (const float*)d_in[7];
  const float* beta  = (const float*)d_in[8];
  float* out = (float*)d_out;

  char* ws = (char*)d_ws;
  size_t off = 0;
  float* featT = (float*)(ws + off);      off += (size_t)BATCH*DHW*CCH*4;   // 256 MB
  float* valT  = (float*)(ws + off);      off += (size_t)BATCH*DHW*CCH*4;   // 256 MB (reused as outT)
  _Float16* cnH    = (_Float16*)(ws + off); off += (size_t)64*64*64*2;
  float*    vcent  = (float*)(ws + off);    off += (size_t)64*64*64*4;
  float*    csum   = (float*)(ws + off);    off += (size_t)64*64*64*4;
  float*    denom  = (float*)(ws + off);    off += (size_t)64*64*4;
  _Float16* cfeatH = (_Float16*)(ws + off); off += (size_t)64*64*64*2;

  (void)hipMemsetAsync(csum,  0, (size_t)64*64*64*4, stream);
  (void)hipMemsetAsync(denom, 0, (size_t)64*64*4,    stream);

  k_qkv  <<<4096, 256, 65536,  stream>>>(x, fw, fb, vw, vb, featT, valT);
  k_pool <<<4096, 64,  0,      stream>>>(featT, valT, cnH, vcent);
  k_passA<<<4096, 256, 180224, stream>>>(featT, valT, cnH, alpha, beta, csum, denom);
  k_cfeat<<<1024, 256, 0,      stream>>>(csum, denom, vcent, cfeatH);
  k_passB<<<4096, 256, 163840, stream>>>(featT, cnH, cfeatH, alpha, beta, valT);
  k_proj <<<4096, 256, 0,      stream>>>(valT, pw, pb, out);
}
